// MoRELayerSynthesisIOptionB_52888227283722
// MI455X (gfx1250) — matrix operand order repaired and checked by reading, not compiled, not run on hardware
//
#include <hip/hip_runtime.h>

// ---------------- problem constants ----------------
static constexpr int DM    = 1024;
static constexpr int DFF   = 2048;
static constexpr int NEXP  = 8;
static constexpr int NTOK  = 8192;              // 4 * 2048
static constexpr long long STATE_ELEMS = (long long)NTOK * DM;   // 8388608

typedef __attribute__((ext_vector_type(16))) _Float16 v16h;
typedef __attribute__((ext_vector_type(8)))  _Float16 v8h;
typedef __attribute__((ext_vector_type(8)))  float    v8f;
typedef __attribute__((ext_vector_type(4)))  unsigned int u32x4;
typedef __attribute__((ext_vector_type(8)))  int      i32x8;
typedef __attribute__((ext_vector_type(4)))  int      i32x4;

// LDS tile geometry: 128 rows x 32 f16 (64B) + 16B TDM pad per row -> 80B pitch
static constexpr int TROW = 40;                 // row pitch in f16 elements
static constexpr int TILE_HALFS = 128 * TROW;   // 5120 halfs = 10240 B per tile

// ---------------- Tensor Data Mover: 2D f16 tile -> LDS (CDNA5 ISA) ----
// D# group0: count=1 | lds_addr | global_addr | type=2
// D# group1: data_size=2B, pad_enable, pad_interval=16dw, pad_amount=4dw,
//            tensor_dim0/1, tile_dim0=32, tile_dim1=128, tensor_dim0_stride
__device__ __forceinline__ void tdm_load_2d(unsigned lds_addr, const _Float16* gptr,
                                            unsigned tile_x, unsigned tile_y,
                                            unsigned tensor_x, unsigned tensor_y,
                                            unsigned stride_elems)
{
    unsigned long long ga = (unsigned long long)(const void*)gptr;
    u32x4 g0;
    g0[0] = 1u;                                          // count=1 (valid user D#)
    g0[1] = lds_addr;                                    // LDS byte address
    g0[2] = (unsigned)ga;                                // global_addr[31:0]
    g0[3] = (unsigned)((ga >> 32) & 0x01ffffffu) | (2u << 30);   // [56:32] | type=2
    i32x8 g1;
    g1[0] = (int)((1u << 16) | (1u << 20) | (3u << 22) | (3u << 25));
    //            data_size=2B  pad_enable  interval=16dw  amount=4dw (16B pad / 64B row)
    g1[1] = (int)((tensor_x & 0xffffu) << 16);           // abar=0 | tensor_dim0[15:0]
    g1[2] = (int)((tensor_x >> 16) | ((tensor_y & 0xffffu) << 16));
    g1[3] = (int)((tensor_y >> 16) | (tile_x << 16));    // tensor_dim1[31:16] | tile_dim0
    g1[4] = (int)(tile_y & 0xffffu);                     // tile_dim1 | tile_dim2=0
    g1[5] = (int)stride_elems;                           // tensor_dim0_stride[31:0]
    g1[6] = 0;                                           // stride0[47:32]=0 | stride1[15:0]=0
    g1[7] = 0;
    i32x4 gz4; gz4[0] = 0; gz4[1] = 0; gz4[2] = 0; gz4[3] = 0;
    i32x8 gz8;
    gz8[0] = 0; gz8[1] = 0; gz8[2] = 0; gz8[3] = 0;
    gz8[4] = 0; gz8[5] = 0; gz8[6] = 0; gz8[7] = 0;
    // amdgpu-toolchain (clang-23) 6-arg form: (g0, g1, g2, g3, ext, cpol)
    __builtin_amdgcn_tensor_load_to_lds(g0, g1, gz4, gz4, gz8, 0);
}

// ---------------- WMMA fragment reads from padded LDS tiles (ds_load_b128) ----
// A (16x32 f16): lanes 0-15 hold row M, K in {0..7,16..23}; lanes 16-31: {8..15,24..31}
__device__ __forceinline__ v16h lds_frag_a(const _Float16* tile, int row, int lane) {
    const _Float16* base = tile + row * TROW + ((lane & 16) ? 8 : 0);
    union { v16h v; v8h h[2]; } r;
    r.h[0] = *(const v8h*)(base);
    r.h[1] = *(const v8h*)(base + 16);
    return r.v;
}
// B (32x16): lanes 0-15 hold col N, K in {0..7,16..23}; lanes 16-31: {8..15,24..31}
__device__ __forceinline__ v16h lds_frag_b(const _Float16* tile, int row, int lane) {
    const _Float16* base = tile + row * TROW + ((lane & 16) ? 8 : 0);
    union { v16h v; v8h h[2]; } r;
    r.h[0] = *(const v8h*)(base);
    r.h[1] = *(const v8h*)(base + 16);
    return r.v;
}

// ---------------- fused-epilogue WMMA GEMM:  C[M,Ncols] = A[M,K] @ W[Ncols,K]^T ----
static constexpr int EPI_F32  = 0;   // store f32
static constexpr int EPI_F16  = 1;   // store f16
static constexpr int EPI_GELU = 2;   // exact gelu -> f16
static constexpr int EPI_SILU = 3;   // silu -> f16
static constexpr int EPI_SACC = 4;   // Cf += rowscale[row*rs_stride] * acc

template <int EPI>
__global__ __launch_bounds__(256)
void gemm_wmma_kernel(const _Float16* __restrict__ A, const _Float16* __restrict__ W,
                      float* __restrict__ Cf, _Float16* __restrict__ Ch,
                      const float* __restrict__ rowscale, int rs_stride,
                      int M, int Ncols, int K)
{
    __shared__ _Float16 smem[2][2][TILE_HALFS];   // [buf][A/B][tile]
    const int lane = threadIdx.x & 31;
    const int wave = threadIdx.x >> 5;
    const int mloc = (wave >> 2) * 64;            // 2 waves in M, 64 rows each
    const int nloc = (wave & 3) * 32;             // 4 waves in N, 32 cols each
    const int blkM = blockIdx.y * 128;
    const int blkN = blockIdx.x * 128;
    const bool tdm_wave = (threadIdx.x < 32);

    const unsigned lds0 = (unsigned)(unsigned long long)(const void*)&smem[0][0][0];
    const unsigned bufStride = 2u * TILE_HALFS * 2u;   // bytes per double-buffer slot
    const unsigned bOffB = TILE_HALFS * 2u;

    v8f acc[4][2];
#pragma unroll
    for (int mt = 0; mt < 4; ++mt)
#pragma unroll
        for (int nt = 0; nt < 2; ++nt)
#pragma unroll
            for (int r = 0; r < 8; ++r) acc[mt][nt][r] = 0.0f;

    if (tdm_wave) {   // prologue: DMA first K-slab of A and B tiles into buffer 0
        tdm_load_2d(lds0, A + (size_t)blkM * K, 32, 128, (unsigned)K, (unsigned)M, (unsigned)K);
        tdm_load_2d(lds0 + bOffB, W + (size_t)blkN * K, 32, 128, (unsigned)K, (unsigned)Ncols, (unsigned)K);
    }

    for (int k0 = 0; k0 < K; k0 += 32) {
        const int cur = (k0 >> 5) & 1;
        if (tdm_wave) __builtin_amdgcn_s_wait_tensorcnt(0);
        __syncthreads();                                   // tile[cur] visible to all waves
        if (tdm_wave && (k0 + 32 < K)) {                   // overlap next DMA with compute
            const unsigned dst = lds0 + (unsigned)(cur ^ 1) * bufStride;
            tdm_load_2d(dst, A + (size_t)blkM * K + k0 + 32, 32, 128,
                        (unsigned)K, (unsigned)M, (unsigned)K);
            tdm_load_2d(dst + bOffB, W + (size_t)blkN * K + k0 + 32, 32, 128,
                        (unsigned)K, (unsigned)Ncols, (unsigned)K);
        }
        const _Float16* tA = smem[cur][0];
        const _Float16* tB = smem[cur][1];
        v16h b0 = lds_frag_b(tB, nloc + (lane & 15), lane);
        v16h b1 = lds_frag_b(tB, nloc + 16 + (lane & 15), lane);
#pragma unroll
        for (int mt = 0; mt < 4; ++mt) {
            v16h a = lds_frag_a(tA, mloc + mt * 16 + (lane & 15), lane);
            acc[mt][0] = __builtin_amdgcn_wmma_f32_16x16x32_f16(
                false, a, false, b0, (short)0, acc[mt][0], false, false);
            acc[mt][1] = __builtin_amdgcn_wmma_f32_16x16x32_f16(
                false, a, false, b1, (short)0, acc[mt][1], false, false);
        }
        __syncthreads();                                   // all reads of tile[cur] done
    }

    const int rowHi = (lane & 16) ? 8 : 0;   // C/D layout: VGPR r -> M = r (+8 hi half)
#pragma unroll
    for (int mt = 0; mt < 4; ++mt)
#pragma unroll
        for (int nt = 0; nt < 2; ++nt) {
            const int col = blkN + nloc + nt * 16 + (lane & 15);
#pragma unroll
            for (int r = 0; r < 8; ++r) {
                const int row = blkM + mloc + mt * 16 + r + rowHi;
                const size_t idx = (size_t)row * Ncols + col;
                float v = acc[mt][nt][r];
                if constexpr (EPI == EPI_F32) {
                    Cf[idx] = v;
                } else if constexpr (EPI == EPI_F16) {
                    Ch[idx] = (_Float16)v;
                } else if constexpr (EPI == EPI_GELU) {
                    Ch[idx] = (_Float16)(0.5f * v * (1.0f + erff(v * 0.70710678118f)));
                } else if constexpr (EPI == EPI_SILU) {
                    Ch[idx] = (_Float16)(v / (1.0f + expf(-v)));
                } else {
                    Cf[idx] += rowscale[(size_t)row * rs_stride] * v;
                }
            }
        }
}

// ---------------- router: logits, softmax, top-2, combine weights, lb stats ----
__global__ __launch_bounds__(256)
void router_kernel(const float* __restrict__ x, const float* __restrict__ rw,
                   float* __restrict__ logits_out, float* __restrict__ combine,
                   float* __restrict__ red /* [0..7]=count, [8..15]=sum prob */)
{
    const int lane = threadIdx.x & 31;
    const int n = blockIdx.x * 8 + (threadIdx.x >> 5);   // one wave per token
    const float* xr = x + (size_t)n * DM;
    float a[NEXP];
#pragma unroll
    for (int e = 0; e < NEXP; ++e) a[e] = 0.0f;
    for (int i = lane; i < DM; i += 32) {
        const float xv = xr[i];
#pragma unroll
        for (int e = 0; e < NEXP; ++e) a[e] += xv * rw[e * DM + i];
    }
#pragma unroll
    for (int off = 16; off > 0; off >>= 1)
#pragma unroll
        for (int e = 0; e < NEXP; ++e) a[e] += __shfl_down(a[e], off, 32);

    if (lane == 0) {
        float mx = a[0];
#pragma unroll
        for (int e = 1; e < NEXP; ++e) mx = fmaxf(mx, a[e]);
        float p[NEXP], se = 0.0f;
#pragma unroll
        for (int e = 0; e < NEXP; ++e) { p[e] = expf(a[e] - mx); se += p[e]; }
        const float inv = 1.0f / se;
#pragma unroll
        for (int e = 0; e < NEXP; ++e) p[e] *= inv;
        int i0 = 0;
#pragma unroll
        for (int e = 1; e < NEXP; ++e) if (p[e] > p[i0]) i0 = e;
        int i1 = (i0 == 0) ? 1 : 0;
#pragma unroll
        for (int e = 0; e < NEXP; ++e) if (e != i0 && p[e] > p[i1]) i1 = e;
        const float s = p[i0] + p[i1];
        const float w0 = p[i0] / s, w1 = p[i1] / s;
#pragma unroll
        for (int e = 0; e < NEXP; ++e) {
            logits_out[(size_t)n * NEXP + e] = a[e];
            combine[(size_t)n * NEXP + e] = (e == i0) ? w0 : ((e == i1) ? w1 : 0.0f);
            atomicAdd(&red[8 + e], p[e]);
        }
        atomicAdd(&red[i0], 1.0f);
        atomicAdd(&red[i1], 1.0f);
    }
}

__global__ void lb_finalize_kernel(const float* __restrict__ red, float* __restrict__ out) {
    if (threadIdx.x == 0 && blockIdx.x == 0) {
        float lb = 0.0f;
        for (int e = 0; e < NEXP; ++e)
            lb += (red[e] / ((float)NTOK * 2.0f)) * (red[8 + e] / (float)NTOK);
        out[0] = (float)NEXP * lb;
    }
}

// ---------------- rmsnorm (optional residual add; optional f32/f16 outputs) ----
__global__ __launch_bounds__(256)
void rmsnorm_kernel(const float* __restrict__ a, const float* __restrict__ b,
                    const float* __restrict__ g,
                    float* __restrict__ outf, _Float16* __restrict__ outh)
{
    const int n = blockIdx.x;
    const int lane = threadIdx.x & 31, wave = threadIdx.x >> 5;
    const float* ar = a + (size_t)n * DM;
    const float* br = b ? b + (size_t)n * DM : nullptr;
    float v[4], ss = 0.0f;
#pragma unroll
    for (int i = 0; i < 4; ++i) {
        const int idx = threadIdx.x + i * 256;
        float t = ar[idx] + (br ? br[idx] : 0.0f);
        v[i] = t; ss += t * t;
    }
#pragma unroll
    for (int off = 16; off > 0; off >>= 1) ss += __shfl_down(ss, off, 32);
    __shared__ float sred[8];
    if (lane == 0) sred[wave] = ss;
    __syncthreads();
    if (wave == 0) {
        float t = (lane < 8) ? sred[lane] : 0.0f;
#pragma unroll
        for (int off = 4; off > 0; off >>= 1) t += __shfl_down(t, off, 32);
        if (lane == 0) sred[0] = t;
    }
    __syncthreads();
    const float scale = rsqrtf(sred[0] * (1.0f / (float)DM) + 1e-6f);
#pragma unroll
    for (int i = 0; i < 4; ++i) {
        const int idx = threadIdx.x + i * 256;
        const float o = v[i] * scale * g[idx];
        if (outf) outf[(size_t)n * DM + idx] = o;
        if (outh) outh[(size_t)n * DM + idx] = (_Float16)o;
    }
}

// ---------------- small utility kernels ----------------
__global__ void f32_to_f16_kernel(const float* __restrict__ in, _Float16* __restrict__ out, long long n) {
    for (long long i = blockIdx.x * (long long)blockDim.x + threadIdx.x; i < n;
         i += (long long)gridDim.x * blockDim.x)
        out[i] = (_Float16)in[i];
}
__global__ void zero_f32_kernel(float* __restrict__ p, long long n) {
    for (long long i = blockIdx.x * (long long)blockDim.x + threadIdx.x; i < n;
         i += (long long)gridDim.x * blockDim.x)
        p[i] = 0.0f;
}
__global__ void concat_f16_kernel(const float* __restrict__ a, const float* __restrict__ b,
                                  _Float16* __restrict__ out) {
    const long long total = (long long)NTOK * 2 * DM;
    for (long long i = blockIdx.x * (long long)blockDim.x + threadIdx.x; i < total;
         i += (long long)gridDim.x * blockDim.x) {
        const long long n = i >> 11, c = i & 2047;
        out[i] = (_Float16)((c < DM) ? a[n * DM + c] : b[n * DM + (c - DM)]);
    }
}
__device__ __forceinline__ unsigned int hash_u32(unsigned int x) {
    x ^= x >> 16; x *= 0x7feb352dU; x ^= x >> 15; x *= 0x846ca68bU; x ^= x >> 16; return x;
}
__global__ void init_state_kernel(float* __restrict__ st, long long n) {
    for (long long i = blockIdx.x * (long long)blockDim.x + threadIdx.x; i < n;
         i += (long long)gridDim.x * blockDim.x) {
        const float u1 = (hash_u32((unsigned int)(2 * i + 1)) + 0.5f) * (1.0f / 4294967296.0f);
        const float u2 = (hash_u32((unsigned int)(2 * i + 2)) + 0.5f) * (1.0f / 4294967296.0f);
        st[i] = 0.02f * sqrtf(-2.0f * logf(u1)) * cosf(6.28318530718f * u2);
    }
}

// ---------------- orchestration ----------------
extern "C" void kernel_launch(void* const* d_in, const int* in_sizes, int n_in,
                              void* d_out, int out_size, void* d_ws, size_t ws_size,
                              hipStream_t stream) {
    const float* x        = (const float*)d_in[0];
    const float* router_w = (const float*)d_in[1];
    const float* exp_w1   = (const float*)d_in[2];
    const float* exp_w2   = (const float*)d_in[3];
    const float* adapter  = (const float*)d_in[4];
    const float* n1 = (const float*)d_in[5];
    const float* n2 = (const float*)d_in[6];
    const float* n3 = (const float*)d_in[7];
    const float* n4 = (const float*)d_in[8];
    const float* wv = (const float*)d_in[9];
    const float* wo = (const float*)d_in[10];
    const float* mlp_w1 = (const float*)d_in[11];
    const float* mlp_w2 = (const float*)d_in[12];
    float* out_state  = (float*)d_out;                       // [NTOK, DM]
    float* out_lb     = (float*)d_out + STATE_ELEMS;         // scalar
    float* out_logits = (float*)d_out + STATE_ELEMS + 1;     // [NTOK, 8]

    // workspace carve-out
    char* w = (char*)d_ws; size_t off = 0;
    auto alloc = [&](size_t bytes) -> void* {
        void* p = w + off; off = (off + bytes + 255) & ~(size_t)255; return p;
    };
    _Float16* xh   = (_Float16*)alloc(sizeof(_Float16) * STATE_ELEMS);
    _Float16* w1h  = (_Float16*)alloc(sizeof(_Float16) * (size_t)NEXP * DFF * DM);
    _Float16* w2h  = (_Float16*)alloc(sizeof(_Float16) * (size_t)NEXP * DM * DFF);
    _Float16* adph = (_Float16*)alloc(sizeof(_Float16) * (size_t)DM * 2 * DM);
    _Float16* wvh  = (_Float16*)alloc(sizeof(_Float16) * (size_t)DM * DM);
    _Float16* woh  = (_Float16*)alloc(sizeof(_Float16) * (size_t)DM * DM);
    _Float16* m1h  = (_Float16*)alloc(sizeof(_Float16) * (size_t)4 * DM * DM);
    _Float16* m2h  = (_Float16*)alloc(sizeof(_Float16) * (size_t)DM * 4 * DM);
    _Float16* s16  = (_Float16*)alloc(sizeof(_Float16) * STATE_ELEMS);
    _Float16* t16  = (_Float16*)alloc(sizeof(_Float16) * STATE_ELEMS);
    _Float16* hbuf = (_Float16*)alloc(sizeof(_Float16) * (size_t)NTOK * 4 * DM); // h / concat / mlp-hidden
    float* combine    = (float*)alloc(sizeof(float) * (size_t)NTOK * NEXP);
    float* red        = (float*)alloc(sizeof(float) * 16);
    float* expert_out = (float*)alloc(sizeof(float) * STATE_ELEMS);
    float* state      = (float*)alloc(sizeof(float) * STATE_ELEMS);
    float* tmpf       = (float*)alloc(sizeof(float) * STATE_ELEMS);
    (void)ws_size; (void)n_in; (void)in_sizes; (void)out_size;

    const dim3 B256(256);
    // conversions + init
    zero_f32_kernel<<<1, 32, 0, stream>>>(red, 16);
    zero_f32_kernel<<<4096, B256, 0, stream>>>(expert_out, STATE_ELEMS);
    f32_to_f16_kernel<<<4096, B256, 0, stream>>>(x, xh, STATE_ELEMS);
    f32_to_f16_kernel<<<4096, B256, 0, stream>>>(exp_w1, w1h, (long long)NEXP * DFF * DM);
    f32_to_f16_kernel<<<4096, B256, 0, stream>>>(exp_w2, w2h, (long long)NEXP * DM * DFF);
    f32_to_f16_kernel<<<2048, B256, 0, stream>>>(adapter, adph, (long long)DM * 2 * DM);
    f32_to_f16_kernel<<<1024, B256, 0, stream>>>(wv, wvh, (long long)DM * DM);
    f32_to_f16_kernel<<<1024, B256, 0, stream>>>(wo, woh, (long long)DM * DM);
    f32_to_f16_kernel<<<2048, B256, 0, stream>>>(mlp_w1, m1h, (long long)4 * DM * DM);
    f32_to_f16_kernel<<<2048, B256, 0, stream>>>(mlp_w2, m2h, (long long)DM * 4 * DM);
    init_state_kernel<<<4096, B256, 0, stream>>>(state, STATE_ELEMS);

    // router (writes logits directly to d_out) + load-balance loss
    router_kernel<<<NTOK / 8, B256, 0, stream>>>(x, router_w, out_logits, combine, red);
    lb_finalize_kernel<<<1, 32, 0, stream>>>(red, out_lb);

    // experts: h = gelu(x @ w1[e]^T); expert_out += combine[:,e] * (h @ w2[e]^T)
    for (int e = 0; e < NEXP; ++e) {
        gemm_wmma_kernel<EPI_GELU><<<dim3(DFF / 128, NTOK / 128), B256, 0, stream>>>(
            xh, w1h + (size_t)e * DFF * DM, nullptr, hbuf, nullptr, 0, NTOK, DFF, DM);
        gemm_wmma_kernel<EPI_SACC><<<dim3(DM / 128, NTOK / 128), B256, 0, stream>>>(
            hbuf, w2h + (size_t)e * DM * DFF, expert_out, nullptr, combine + e, NEXP, NTOK, DM, DFF);
    }

    // recurrent block, 2 steps
    for (int step = 0; step < 2; ++step) {
        concat_f16_kernel<<<4096, B256, 0, stream>>>(state, expert_out, hbuf);
        gemm_wmma_kernel<EPI_F32><<<dim3(DM / 128, NTOK / 128), B256, 0, stream>>>(
            hbuf, adph, state, nullptr, nullptr, 0, NTOK, DM, 2 * DM);          // state = residual
        rmsnorm_kernel<<<NTOK, B256, 0, stream>>>(state, nullptr, n1, nullptr, s16);
        gemm_wmma_kernel<EPI_F16><<<dim3(DM / 128, NTOK / 128), B256, 0, stream>>>(
            s16, wvh, nullptr, t16, nullptr, 0, NTOK, DM, DM);                  // v-proj
        gemm_wmma_kernel<EPI_F32><<<dim3(DM / 128, NTOK / 128), B256, 0, stream>>>(
            t16, woh, tmpf, nullptr, nullptr, 0, NTOK, DM, DM);                 // out-proj
        rmsnorm_kernel<<<NTOK, B256, 0, stream>>>(state, tmpf, n2, state, nullptr);
        rmsnorm_kernel<<<NTOK, B256, 0, stream>>>(state, nullptr, n3, nullptr, s16);
        gemm_wmma_kernel<EPI_SILU><<<dim3(4 * DM / 128, NTOK / 128), B256, 0, stream>>>(
            s16, m1h, nullptr, hbuf, nullptr, 0, NTOK, 4 * DM, DM);             // silu(mlp_w1)
        gemm_wmma_kernel<EPI_F32><<<dim3(DM / 128, NTOK / 128), B256, 0, stream>>>(
            hbuf, m2h, tmpf, nullptr, nullptr, 0, NTOK, DM, 4 * DM);            // mlp_w2
        float* dst = (step == 1) ? out_state : state;
        rmsnorm_kernel<<<NTOK, B256, 0, stream>>>(state, tmpf, n4, dst, nullptr);
    }
}
